// AgentMatchingDecoder_31671088841418
// MI455X (gfx1250) — matrix operand order repaired and checked by reading, not compiled, not run on hardware
//
#include <hip/hip_runtime.h>
#include <hip/hip_bf16.h>

typedef __attribute__((ext_vector_type(16))) _Float16 v16h;
typedef __attribute__((ext_vector_type(8)))  float    v8f;

#define BSZ 16
#define NAG 256
#define HWN 1024
#define CHN 512
#define C3N 1536

// ---------------------------------------------------------------- conversions
__global__ __launch_bounds__(256)
void cvt_f32_f16(const float* __restrict__ in, _Float16* __restrict__ out, long long n) {
    long long i = (long long)blockIdx.x * 256 + threadIdx.x;
    if (i < n) out[i] = (_Float16)in[i];
}

// ---------------------------------------------------------------- WMMA GEMM
// C[m,n] = scale * sum_k A[m,k]*B[k,n] (+bias[n]) (relu)
// TA: A stored [K,M] row-major (lda = M). TB: B stored [N,K] row-major (ldb = K).
// Block = 256 threads (8 waves), tile 128x64, BK = 32, double-buffered LDS.
// Wave tile 32x32: 2 A-frags x 2 B-frags -> 4 WMMAs per K-step.
// Non-transposed tile loads use GLOBAL_LOAD_ASYNC_TO_LDS_B128 (ASYNCcnt),
// transposed ones use sync load + 16-bit LDS scatter (+ global_prefetch).
template<bool TA, bool TB, bool HAS_BIAS, bool RELU, bool W32, bool W16>
__global__ __launch_bounds__(256)
void gemm_wmma(const _Float16* __restrict__ A, long long sA,
               const _Float16* __restrict__ B, long long sB,
               const float* __restrict__ bias,
               float* __restrict__ C32,
               _Float16* __restrict__ C16,
               int M, int N, int K, float scale)
{
    __shared__ _Float16 As[2][128][40];  // [M][K], row stride 80B (16B aligned)
    __shared__ _Float16 Bs[2][32][72];   // [K][N], row stride 144B (16B aligned)

    const int tid = threadIdx.x;
    const int bx  = blockIdx.x * 64;         // n0
    const int by  = blockIdx.y * 128;        // m0
    A += (long long)blockIdx.z * sA;
    B += (long long)blockIdx.z * sB;

    const int wave = tid >> 5;
    const int lane = tid & 31;
    const int wm   = (wave & 3) << 5;        // wave row offset (4 waves x 32 rows)
    const int wn   = (wave >> 2) << 5;       // wave col offset (2 waves x 32 cols)
    const int lrow = lane & 15;
    const int g    = lane >> 4;

    v8f acc00 = {}, acc01 = {}, acc10 = {}, acc11 = {};

    // async-to-LDS issues per wave per tile (A needs 2 chunks/thread, B needs 1)
    constexpr int NISSUE = (TA ? 0 : 2) + (TB ? 0 : 1);

    auto load_tile = [&](int k0, int buf) {
        // ---- A tile (128x32) -> LDS[buf]
        if (!TA) {
            int m  = tid >> 2;               // 0..63
            int kk = (tid & 3) << 3;         // 0,8,16,24
            const _Float16* s0 = A + (long long)(by + m) * K + k0 + kk;
            const _Float16* s1 = A + (long long)(by + m + 64) * K + k0 + kk;
            unsigned d0 = (unsigned)(unsigned long long)&As[buf][m][kk];
            unsigned d1 = (unsigned)(unsigned long long)&As[buf][m + 64][kk];
            asm volatile("global_load_async_to_lds_b128 %0, %1, off"
                         :: "v"(d0), "v"((unsigned long long)s0) : "memory");
            asm volatile("global_load_async_to_lds_b128 %0, %1, off"
                         :: "v"(d1), "v"((unsigned long long)s1) : "memory");
        } else {
            int kk = tid >> 4;               // 0..15
            int m  = (tid & 15) << 3;        // 0..120
            __builtin_prefetch(A + (long long)(k0 + 32 + kk) * M + by + m, 0, 0);
            #pragma unroll
            for (int c = 0; c < 2; ++c) {
                int kc = kk + (c << 4);      // kk, kk+16
                union { uint4 q; _Float16 h[8]; } t;
                t.q = *(const uint4*)(A + (long long)(k0 + kc) * M + by + m);
                #pragma unroll
                for (int j = 0; j < 8; ++j) As[buf][m + j][kc] = t.h[j];
            }
        }
        // ---- B tile (32x64) -> LDS[buf]
        if (!TB) {
            int kk = tid >> 3;               // 0..31
            int n  = (tid & 7) << 3;         // 0..56
            const _Float16* src = B + (long long)(k0 + kk) * N + bx + n;
            unsigned dst = (unsigned)(unsigned long long)&Bs[buf][kk][n];
            asm volatile("global_load_async_to_lds_b128 %0, %1, off"
                         :: "v"(dst), "v"((unsigned long long)src) : "memory");
        } else {
            int n  = tid >> 2;               // 0..63
            int kk = (tid & 3) << 3;         // 0,8,16,24
            __builtin_prefetch(B + (long long)(bx + n) * K + k0 + 32 + kk, 0, 0);
            union { uint4 q; _Float16 h[8]; } t;
            t.q = *(const uint4*)(B + (long long)(bx + n) * K + k0 + kk);
            #pragma unroll
            for (int j = 0; j < 8; ++j) Bs[buf][kk + j][n] = t.h[j];
        }
    };

    auto wait_async = [&](bool steady) {
        if (NISSUE == 3) {
            if (steady) asm volatile("s_wait_asynccnt 3" ::: "memory");
            else        asm volatile("s_wait_asynccnt 0" ::: "memory");
        } else if (NISSUE == 2) {
            if (steady) asm volatile("s_wait_asynccnt 2" ::: "memory");
            else        asm volatile("s_wait_asynccnt 0" ::: "memory");
        } else if (NISSUE == 1) {
            if (steady) asm volatile("s_wait_asynccnt 1" ::: "memory");
            else        asm volatile("s_wait_asynccnt 0" ::: "memory");
        }
    };

    const int nk = K >> 5;
    load_tile(0, 0);
    int cur = 0;
    for (int t = 0; t < nk; ++t) {
        const bool more = (t + 1 < nk);            // uniform
        if (more) load_tile((t + 1) << 5, cur ^ 1);
        wait_async(more);       // my tile-t async copies have landed in LDS
        __syncthreads();        // everyone's tile-t data visible

        // ---- A fragments: lane holds row M=lrow (+16), K per ISA 16-bit A layout
        union { v16h v; unsigned int u[8]; } af0, af1;
        #pragma unroll
        for (int p = 0; p < 8; ++p) {
            int h  = p << 1;
            int kk = ((h & 8) << 1) + (h & 7) + (g << 3);
            af0.u[p] = *(const unsigned int*)&As[cur][wm + lrow][kk];
            af1.u[p] = *(const unsigned int*)&As[cur][wm + 16 + lrow][kk];
        }
        // ---- B fragments: lane holds N=lrow (+16), K per ISA 16-bit B layout
        union { v16h v; _Float16 h[16]; } bf0, bf1;
        #pragma unroll
        for (int s = 0; s < 16; ++s) {
            int kk = ((s & 8) << 1) + (s & 7) + (g << 3);
            bf0.h[s] = Bs[cur][kk][wn + lrow];
            bf1.h[s] = Bs[cur][kk][wn + 16 + lrow];
        }

        acc00 = __builtin_amdgcn_wmma_f32_16x16x32_f16(false, af0.v, false, bf0.v,
                                                       (short)0, acc00, false, false);
        acc01 = __builtin_amdgcn_wmma_f32_16x16x32_f16(false, af0.v, false, bf1.v,
                                                       (short)0, acc01, false, false);
        acc10 = __builtin_amdgcn_wmma_f32_16x16x32_f16(false, af1.v, false, bf0.v,
                                                       (short)0, acc10, false, false);
        acc11 = __builtin_amdgcn_wmma_f32_16x16x32_f16(false, af1.v, false, bf1.v,
                                                       (short)0, acc11, false, false);
        __syncthreads();        // done reading buf[cur]; next iter may overwrite it
        cur ^= 1;
    }

    // ---- epilogue: C layout M = r + 8*g, N = lrow (+16 for frag1)
    const long long cbase = (long long)blockIdx.z * (long long)M * N;
    const int n0 = bx + wn + lrow;
    float bias0 = HAS_BIAS ? bias[n0]      : 0.0f;
    float bias1 = HAS_BIAS ? bias[n0 + 16] : 0.0f;

    auto store_pair = [&](v8f& a0, v8f& a1, int mOff) {
        #pragma unroll
        for (int r = 0; r < 8; ++r) {
            int m = by + wm + mOff + r + (g << 3);
            float v0 = a0[r] * scale + bias0;
            float v1 = a1[r] * scale + bias1;
            if (RELU) { v0 = v0 > 0.0f ? v0 : 0.0f; v1 = v1 > 0.0f ? v1 : 0.0f; }
            long long o0 = cbase + (long long)m * N + n0;
            if (W32) { C32[o0] = v0; C32[o0 + 16] = v1; }
            if (W16) { C16[o0] = (_Float16)v0; C16[o0 + 16] = (_Float16)v1; }
        }
    };
    store_pair(acc00, acc01, 0);
    store_pair(acc10, acc11, 16);
}

// ---------------------------------------------------------------- argmax
// idx[b,p] = argmax_a S[b*batchStride + p*posStride + a*aStride], first-max tie-break
__global__ __launch_bounds__(256)
void argmax_k(const float* __restrict__ S, int* __restrict__ idx,
              int count, long long batchStride, long long aStride,
              long long posStride, int npos)
{
    int t = blockIdx.x * 256 + threadIdx.x;
    int b = t / npos, p = t % npos;
    const float* base = S + (long long)b * batchStride + (long long)p * posStride;
    float best = base[0];
    int   bi   = 0;
    for (int a = 1; a < count; ++a) {
        float v = base[(long long)a * aStride];
        if (v > best) { best = v; bi = a; }
    }
    idx[t] = bi;
}

// ---------------------------------------------------------------- masked softmax
// probs[b,i,j] = softmax_j( logits[b,i,j] + (a_idx[b,i]==b_idx[b,j] ? 0 : -1e6) )
__global__ __launch_bounds__(256)
void softmax_align_k(const float* __restrict__ logits, const int* __restrict__ aidx,
                     const int* __restrict__ bidx, _Float16* __restrict__ probs)
{
    __shared__ float red[256];
    const int i = blockIdx.x, b = blockIdx.y, tid = threadIdx.x;
    const long long rowoff = ((long long)b * HWN + i) * HWN;
    const float* row = logits + rowoff;
    const int*   bb  = bidx + b * HWN;
    const int    ai  = aidx[b * HWN + i];

    float x[4];
    float lmax = -3.0e38f;
    #pragma unroll
    for (int t = 0; t < 4; ++t) {
        int j = tid + t * 256;
        float v = row[j] + ((bb[j] == ai) ? 0.0f : -1.0e6f);
        x[t] = v;
        lmax = fmaxf(lmax, v);
    }
    red[tid] = lmax; __syncthreads();
    for (int s = 128; s > 0; s >>= 1) {
        if (tid < s) red[tid] = fmaxf(red[tid], red[tid + s]);
        __syncthreads();
    }
    float m = red[0]; __syncthreads();

    float lsum = 0.0f;
    #pragma unroll
    for (int t = 0; t < 4; ++t) { x[t] = __expf(x[t] - m); lsum += x[t]; }
    red[tid] = lsum; __syncthreads();
    for (int s = 128; s > 0; s >>= 1) {
        if (tid < s) red[tid] += red[tid + s];
        __syncthreads();
    }
    float inv = 1.0f / red[0];
    #pragma unroll
    for (int t = 0; t < 4; ++t)
        probs[rowoff + tid + t * 256] = (_Float16)(x[t] * inv);
}

// ---------------------------------------------------------------- host
extern "C" void kernel_launch(void* const* d_in, const int* in_sizes, int n_in,
                              void* d_out, int out_size, void* d_ws, size_t ws_size,
                              hipStream_t stream)
{
    (void)in_sizes; (void)n_in; (void)out_size; (void)ws_size;

    const float* tok   = (const float*)d_in[0];
    const float* supp  = (const float*)d_in[1];
    const float* query = (const float*)d_in[2];
    const float* W_qa  = (const float*)d_in[3];  const float* b_qa = (const float*)d_in[4];
    const float* W_ks  = (const float*)d_in[5];  const float* b_ks = (const float*)d_in[6];
    const float* W_ka  = (const float*)d_in[7];  const float* b_ka = (const float*)d_in[8];
    const float* W_vs  = (const float*)d_in[9];  const float* b_vs = (const float*)d_in[10];
    const float* W_f1  = (const float*)d_in[11]; const float* b_f1 = (const float*)d_in[12];
    const float* W_f2  = (const float*)d_in[13]; const float* b_f2 = (const float*)d_in[14];
    float* out = (float*)d_out;

    char* ws = (char*)d_ws;
    const size_t MB = 1ull << 20;
    // --- POOL_A (76MB), live through the score GEMMs / argmax ---
    _Float16* tokA_h  = (_Float16*)(ws + 0 * MB);    //  4MB
    _Float16* supp_h  = (_Float16*)(ws + 4 * MB);    // 16MB
    _Float16* query_h = (_Float16*)(ws + 20 * MB);   // 16MB
    _Float16* qa_h    = (_Float16*)(ws + 36 * MB);   //  4MB
    _Float16* ka_h    = (_Float16*)(ws + 40 * MB);   //  4MB
    float*    sAs32   = (float*)   (ws + 44 * MB);   // 16MB
    float*    sQa32   = (float*)   (ws + 60 * MB);   // 16MB  (pool ends @76MB)
    float*    logits  = (float*)   (ws + 0 * MB);    // 64MB, alias: live only logits->softmax
    _Float16* hid_h   = (_Float16*)(ws + 0 * MB);    // 48MB, alias: live only ffn1->ffn2
    // --- persistent-ish region ---
    _Float16* ks_h    = (_Float16*)(ws + 76 * MB);   // 16MB
    _Float16* qq_h    = (_Float16*)(ws + 92 * MB);   // 16MB
    _Float16* probs_h = (_Float16*)(ws + 76 * MB);   // 32MB alias of ks_h+qq_h (dead by then)
    _Float16* vs_h    = (_Float16*)(ws + 108 * MB);  // 16MB
    _Float16* sAs16   = (_Float16*)(ws + 124 * MB);  //  8MB
    _Float16* sQa16   = (_Float16*)(ws + 132 * MB);  //  8MB
    _Float16* dec_h   = (_Float16*)(ws + 124 * MB);  // 16MB alias of sAs16+sQa16
    _Float16* Wqa_h   = (_Float16*)(ws + 140 * MB);
    _Float16* Wks_h   = (_Float16*)(ws + 140 * MB + 512 * 1024);
    _Float16* Wka_h   = (_Float16*)(ws + 141 * MB);
    _Float16* Wvs_h   = (_Float16*)(ws + 141 * MB + 512 * 1024);
    _Float16* Wf1_h   = (_Float16*)(ws + 142 * MB);
    _Float16* Wf2_h   = (_Float16*)(ws + 142 * MB + 1536 * 1024);
    int*      aidx    = (int*)     (ws + 145 * MB);
    int*      bidx    = (int*)     (ws + 145 * MB + 64 * 1024);

    const long long nTok = (long long)BSZ * NAG * CHN;   // 2,097,152
    const long long nHWC = (long long)BSZ * HWN * CHN;   // 8,388,608
    const long long nW   = (long long)CHN * CHN;         //   262,144
    const long long nWf  = (long long)CHN * C3N;         //   786,432

    // ---- fp32 -> fp16 conversions
    cvt_f32_f16<<<(unsigned)(nTok / 256), 256, 0, stream>>>(tok,   tokA_h,  nTok);
    cvt_f32_f16<<<(unsigned)(nHWC / 256), 256, 0, stream>>>(supp,  supp_h,  nHWC);
    cvt_f32_f16<<<(unsigned)(nHWC / 256), 256, 0, stream>>>(query, query_h, nHWC);
    cvt_f32_f16<<<(unsigned)(nW   / 256), 256, 0, stream>>>(W_qa, Wqa_h, nW);
    cvt_f32_f16<<<(unsigned)(nW   / 256), 256, 0, stream>>>(W_ks, Wks_h, nW);
    cvt_f32_f16<<<(unsigned)(nW   / 256), 256, 0, stream>>>(W_ka, Wka_h, nW);
    cvt_f32_f16<<<(unsigned)(nW   / 256), 256, 0, stream>>>(W_vs, Wvs_h, nW);
    cvt_f32_f16<<<(unsigned)(nWf  / 256), 256, 0, stream>>>(W_f1, Wf1_h, nWf);
    cvt_f32_f16<<<(unsigned)(nWf  / 256), 256, 0, stream>>>(W_f2, Wf2_h, nWf);

    // ---- projections (A batched, B = shared weights)
    gemm_wmma<false,false,true,false,false,true><<<dim3(CHN/64, NAG/128, BSZ), 256, 0, stream>>>(
        tokA_h, (long long)NAG*CHN, Wqa_h, 0, b_qa, nullptr, qa_h, NAG, CHN, CHN, 1.0f);
    gemm_wmma<false,false,true,false,false,true><<<dim3(CHN/64, HWN/128, BSZ), 256, 0, stream>>>(
        supp_h, (long long)HWN*CHN, Wks_h, 0, b_ks, nullptr, ks_h, HWN, CHN, CHN, 1.0f);
    gemm_wmma<false,false,true,false,false,true><<<dim3(CHN/64, HWN/128, BSZ), 256, 0, stream>>>(
        query_h, (long long)HWN*CHN, Wqa_h, 0, b_qa, nullptr, qq_h, HWN, CHN, CHN, 1.0f);
    gemm_wmma<false,false,true,false,false,true><<<dim3(CHN/64, NAG/128, BSZ), 256, 0, stream>>>(
        tokA_h, (long long)NAG*CHN, Wka_h, 0, b_ka, nullptr, ka_h, NAG, CHN, CHN, 1.0f);
    gemm_wmma<false,false,true,false,false,true><<<dim3(CHN/64, HWN/128, BSZ), 256, 0, stream>>>(
        supp_h, (long long)HWN*CHN, Wvs_h, 0, b_vs, nullptr, vs_h, HWN, CHN, CHN, 1.0f);

    // ---- scores (B = batched activations, transposed; scale = 1/sqrt(64))
    gemm_wmma<false,true,false,false,true,true><<<dim3(HWN/64, NAG/128, BSZ), 256, 0, stream>>>(
        qa_h, (long long)NAG*CHN, ks_h, (long long)HWN*CHN, nullptr,
        sAs32, sAs16, NAG, HWN, CHN, 0.125f);
    gemm_wmma<false,true,false,false,true,true><<<dim3(NAG/64, HWN/128, BSZ), 256, 0, stream>>>(
        qq_h, (long long)HWN*CHN, ka_h, (long long)NAG*CHN, nullptr,
        sQa32, sQa16, HWN, NAG, CHN, 0.125f);

    // ---- argmaxes
    argmax_k<<<(BSZ*HWN)/256, 256, 0, stream>>>(sAs32, aidx, NAG,
        (long long)NAG*HWN, (long long)HWN, 1LL, HWN);
    argmax_k<<<(BSZ*HWN)/256, 256, 0, stream>>>(sQa32, bidx, NAG,
        (long long)HWN*NAG, 1LL, (long long)NAG, HWN);

    // ---- logits[i,j] = sum_a sAs[a,i]*sQa[j,a]  (A^T, B^T), K = NA
    gemm_wmma<true,true,false,false,true,false><<<dim3(HWN/64, HWN/128, BSZ), 256, 0, stream>>>(
        sAs16, (long long)NAG*HWN, sQa16, (long long)HWN*NAG, nullptr,
        logits, nullptr, HWN, HWN, NAG, 1.0f);

    // ---- masked softmax over j
    softmax_align_k<<<dim3(HWN, BSZ), 256, 0, stream>>>(logits, aidx, bidx, probs_h);

    // ---- dec[j,c] = sum_i P[i,j]*vs[i,c]  (A^T), K = HW
    gemm_wmma<true,false,false,false,false,true><<<dim3(CHN/64, HWN/128, BSZ), 256, 0, stream>>>(
        probs_h, (long long)HWN*HWN, vs_h, (long long)HWN*CHN, nullptr,
        nullptr, dec_h, HWN, CHN, HWN, 1.0f);

    // ---- FFN
    gemm_wmma<false,false,true,true,false,true><<<dim3(C3N/64, HWN/128, BSZ), 256, 0, stream>>>(
        dec_h, (long long)HWN*CHN, Wf1_h, 0, b_f1, nullptr, hid_h, HWN, C3N, CHN, 1.0f);
    gemm_wmma<false,false,true,false,true,false><<<dim3(CHN/64, HWN/128, BSZ), 256, 0, stream>>>(
        hid_h, (long long)HWN*C3N, Wf2_h, 0, b_f2, out, nullptr, HWN, CHN, C3N, 1.0f);
}
